// SoftRegion_55293408969027
// MI455X (gfx1250) — matrix operand order repaired and checked by reading, not compiled, not run on hardware
//
#include <hip/hip_runtime.h>
#include <stdint.h>

// ---------------------------------------------------------------------------
// Problem constants (fixed by the reference)
// ---------------------------------------------------------------------------
#define BB 256
#define LL 196
#define CC 768
#define KK 2048
#define BL (BB * LL)            // 50176 tokens
#define NCHUNK (CC / 32)        // 24 k-chunks of 32 for bf16 WMMA
#define NTILE (KK / 16)         // 128 column tiles of 16 codes
#define ROWS_PER_BLOCK 128      // 8 waves * 16 rows
#define TILE_SHORTS (NCHUNK * 32 * 16)  // 12288 bf16 per packed B tile (24 KB)

typedef __bf16 bf16_t;
typedef bf16_t v16bf __attribute__((ext_vector_type(16)));
typedef float  v8f   __attribute__((ext_vector_type(8)));
typedef int    v4i_vs __attribute__((__vector_size__(16)));   // int4 vector

// CDNA5 async global->LDS DMA path (ASYNCcnt), if this toolchain exposes it.
#if defined(__has_builtin)
#if __has_builtin(__builtin_amdgcn_global_load_async_to_lds_b128)
#define HAVE_ASYNC_LDS 1
#endif
#endif
#ifndef HAVE_ASYNC_LDS
#define HAVE_ASYNC_LDS 0
#endif

__device__ __forceinline__ void wait_async0() {
#if __has_builtin(__builtin_amdgcn_s_wait_asynccnt)
  __builtin_amdgcn_s_wait_asynccnt(0);
#else
  asm volatile("s_wait_asynccnt 0" ::: "memory");
#endif
}

#if HAVE_ASYNC_LDS
// Builtin signature (from hipcc diagnostics): param0 = int4* in global (AS1),
// param1 = int4* in LDS (AS3), then immediate offset + cpol. Offset applies to
// BOTH addresses, which is
// exactly what we want for strided cooperative copies.
template <int OFF>
__device__ __forceinline__ void async_ld128(const void* g, void* l) {
  __builtin_amdgcn_global_load_async_to_lds_b128(
      (__attribute__((address_space(1))) v4i_vs*)g,
      (__attribute__((address_space(3))) v4i_vs*)l, OFF, 0);
}
// Copy 6 x 16B per thread (256 threads -> 24 KB tile), 4096B slice stride.
__device__ __forceinline__ void async_copy_tile(const uint16_t* g, uint16_t* l) {
  async_ld128<0>(g, l);
  async_ld128<4096>(g, l);
  async_ld128<8192>(g, l);
  async_ld128<12288>(g, l);
  async_ld128<16384>(g, l);
  async_ld128<20480>(g, l);
}
#endif

// float -> bf16 (round to nearest even)
__device__ __forceinline__ uint16_t f2bf_bits(float f) {
  union { float f; uint32_t u; } v; v.f = f;
  uint32_t r = v.u + 0x7FFFu + ((v.u >> 16) & 1u);
  return (uint16_t)(r >> 16);
}
__device__ __forceinline__ bf16_t f2bf(float f) {
  uint16_t b = f2bf_bits(f);
  bf16_t o; __builtin_memcpy(&o, &b, 2);
  return o;
}

// ---------------------------------------------------------------------------
// Kernel 1: halfnorm[k] = 0.5 * ||codebook[k]||^2
// ---------------------------------------------------------------------------
__global__ __launch_bounds__(256) void halfnorm_kernel(const float* __restrict__ CB,
                                                       float* __restrict__ hn) {
  __shared__ float red[256];
  const int k = blockIdx.x;
  const int tid = threadIdx.x;
  const float* row = CB + (size_t)k * CC;
  float s = 0.f;
  for (int c = tid; c < CC; c += 256) { float v = row[c]; s += v * v; }
  red[tid] = s;
  __syncthreads();
  for (int st = 128; st > 0; st >>= 1) {
    if (tid < st) red[tid] += red[tid + st];
    __syncthreads();
  }
  if (tid == 0) hn[k] = 0.5f * red[0];
}

// ---------------------------------------------------------------------------
// Kernel 2: pack codebook into WMMA B-fragment layout (bf16).
// B tile (K=32 x N=16) for column tile `tile`, k-chunk `chunk`:
//   lane l holds n = l%16; element s is k = chunk*32 + 8*(l/16) + s (s<8),
//   k = chunk*32 + 16 + 8*(l/16) + (s-8) (s>=8).
// Packed layout: P[tile][chunk][lane][16 bf16] (32B per lane, contiguous).
// ---------------------------------------------------------------------------
__global__ __launch_bounds__(32) void pack_kernel(const float* __restrict__ CB,
                                                  uint16_t* __restrict__ P) {
  const int blk  = blockIdx.x;
  const int lane = threadIdx.x;          // 0..31
  const int tile  = blk / NCHUNK;
  const int chunk = blk % NCHUNK;
  const int k  = chunk * 32 + (lane >> 4) * 8;
  const int n0 = tile * 16;
  uint16_t tmp[16];
#pragma unroll
  for (int s = 0; s < 16; ++s)
    tmp[s] = f2bf_bits(CB[(size_t)(n0 + (lane & 15)) * CC + k + (s < 8 ? s : s + 8)]);
  uint16_t* dst = P + ((size_t)(tile * NCHUNK + chunk) * 32 + lane) * 16;
  __builtin_memcpy(dst, tmp, 32);        // two b128 stores
}

// ---------------------------------------------------------------------------
// Kernel 3: the WMMA VQ-argmax kernel.
// grid = 392 blocks of 256 threads (8 waves); wave w owns 16 rows of X.
// A fragments (16x32 bf16 per chunk) live in registers for the whole kernel;
// B tiles stream through double-buffered LDS shared by all 8 waves, fed by
// GLOBAL_LOAD_ASYNC_TO_LDS_B128 DMA (ASYNCcnt) when available.
// score = x.e - 0.5||e||^2 ; argmax over 2048 codes == argmin distance.
// ---------------------------------------------------------------------------
__global__ __launch_bounds__(256, 1) void vq_argmax_kernel(const float* __restrict__ X,
                                                           const uint16_t* __restrict__ P,
                                                           const float* __restrict__ halfnorm,
                                                           int* __restrict__ idx_out) {
  __shared__ __align__(32) uint16_t sB[2][TILE_SHORTS];   // 2 x 24 KB
  __shared__ float sHN[KK];                               // 8 KB

  const int tid  = threadIdx.x;
  const int lane = tid & 31;
  const int wave = tid >> 5;
  const int lcol = lane & 15;            // column-within-tile this lane owns (C/D N index)
  const int hi   = lane >> 4;

  // ---- load this wave's A tile (16 rows x 768) into bf16 fragments -------
  // A layout (16-bit 16x32): lane m; elements 0..7 -> k = hi*8 + 0..7,
  // elements 8..15 -> k = 16 + hi*8 + 0..7.
  const int row = blockIdx.x * ROWS_PER_BLOCK + wave * 16 + lcol;
  const float* xr = X + (size_t)row * CC;
  v16bf afrag[NCHUNK];
#pragma unroll
  for (int c = 0; c < NCHUNK; ++c) {
    const float4* p0 = (const float4*)(xr + c * 32 + hi * 8);
    const float4* p1 = (const float4*)(xr + c * 32 + 16 + hi * 8);
    float4 u0 = p0[0], u1 = p0[1], v0 = p1[0], v1 = p1[1];
    v16bf a;
    a[0] = f2bf(u0.x);  a[1] = f2bf(u0.y);  a[2] = f2bf(u0.z);  a[3] = f2bf(u0.w);
    a[4] = f2bf(u1.x);  a[5] = f2bf(u1.y);  a[6] = f2bf(u1.z);  a[7] = f2bf(u1.w);
    a[8] = f2bf(v0.x);  a[9] = f2bf(v0.y);  a[10] = f2bf(v0.z); a[11] = f2bf(v0.w);
    a[12] = f2bf(v1.x); a[13] = f2bf(v1.y); a[14] = f2bf(v1.z); a[15] = f2bf(v1.w);
    afrag[c] = a;
  }

  // Per-thread base pointers for the cooperative 24 KB tile copy:
  // 1536 x 16B per tile, 6 x 16B per thread, 4096B stride between slices.
  const uint16_t* gbase = P + (size_t)tid * 8;

  // ---- prologue: stage tile 0 and the half-norms into LDS ----------------
#if HAVE_ASYNC_LDS
  {
    async_copy_tile(gbase, &sB[0][tid * 8]);
    for (int i = tid; i < KK; i += 256) sHN[i] = halfnorm[i];
    wait_async0();
  }
#else
  {
    const uint4* s0 = (const uint4*)P;
    uint4* d0 = (uint4*)&sB[0][0];
#pragma unroll
    for (int j = 0; j < 6; ++j) d0[tid + j * 256] = s0[tid + j * 256];
    for (int i = tid; i < KK; i += 256) sHN[i] = halfnorm[i];
  }
#endif
  __syncthreads();

  float bestv[8];
  int   besti[8];
#pragma unroll
  for (int r = 0; r < 8; ++r) { bestv[r] = -3.0e38f; besti[r] = 0; }

  // ---- main loop over 128 column tiles -----------------------------------
  for (int t = 0; t < NTILE; ++t) {
    const int cb = t & 1;
    const bool has_next = (t + 1) < NTILE;

#if HAVE_ASYNC_LDS
    // kick off DMA of the next packed B tile into the other LDS buffer
    if (has_next)
      async_copy_tile(gbase + (size_t)(t + 1) * TILE_SHORTS,
                      &sB[(t + 1) & 1][tid * 8]);
#else
    uint4 stg[6];
    if (has_next) {
      const uint4* nsrc = (const uint4*)(P + (size_t)(t + 1) * TILE_SHORTS);
#pragma unroll
      for (int j = 0; j < 6; ++j) stg[j] = nsrc[tid + j * 256];
    }
#endif

    // 24 chained bf16 WMMAs, split over two accumulators to break the RAW chain
    v8f acc0 = {}; v8f acc1 = {};
    const uint16_t* bbase = &sB[cb][0];
#pragma unroll
    for (int c = 0; c < NCHUNK; c += 2) {
      v16bf b0 = *(const v16bf*)(bbase + ((c + 0) * 32 + lane) * 16);
      acc0 = __builtin_amdgcn_wmma_f32_16x16x32_bf16(false, afrag[c + 0], false, b0,
                                                     (short)0, acc0, false, false);
      v16bf b1 = *(const v16bf*)(bbase + ((c + 1) * 32 + lane) * 16);
      acc1 = __builtin_amdgcn_wmma_f32_16x16x32_bf16(false, afrag[c + 1], false, b1,
                                                     (short)0, acc1, false, false);
    }

    // score = dot - 0.5||e||^2 ; update per-lane running argmax
    const float hn  = sHN[t * 16 + lcol];
    const int   col = t * 16 + lcol;
#pragma unroll
    for (int r = 0; r < 8; ++r) {
      float s = (acc0[r] + acc1[r]) - hn;
      if (s > bestv[r]) { bestv[r] = s; besti[r] = col; }  // strict > keeps first idx
    }

#if HAVE_ASYNC_LDS
    if (has_next) wait_async0();     // DMA for tile t+1 done before anyone reads it
#else
    if (has_next) {
      uint4* nd = (uint4*)&sB[(t + 1) & 1][0];
#pragma unroll
      for (int j = 0; j < 6; ++j) nd[tid + j * 256] = stg[j];
    }
#endif
    __syncthreads();
  }

  // ---- reduce argmax across the 16 lanes holding each row ----------------
  // C/D layout: lanes 0-15 hold rows r (VGPR r), lanes 16-31 hold rows r+8.
#pragma unroll
  for (int r = 0; r < 8; ++r) {
    float bv = bestv[r]; int bi = besti[r];
#pragma unroll
    for (int off = 8; off >= 1; off >>= 1) {
      float ov = __shfl_xor(bv, off, 16);
      int   oi = __shfl_xor(bi, off, 16);
      if (ov > bv || (ov == bv && oi < bi)) { bv = ov; bi = oi; }
    }
    bestv[r] = bv; besti[r] = bi;
  }
  const int r = lcol;
  if (r < 8) {
    int sel = besti[0];
#pragma unroll
    for (int rr = 1; rr < 8; ++rr) if (r == rr) sel = besti[rr];
    const int m = (hi << 3) + r;   // lanes 0-7 -> rows 0-7, lanes 16-23 -> rows 8-15
    idx_out[blockIdx.x * ROWS_PER_BLOCK + wave * 16 + m] = sel;
  }
}

// ---------------------------------------------------------------------------
// Kernel 4: att[i] = relu( codebook[idx[i]] . att_w[0,:,1] )  (wave per token)
// ---------------------------------------------------------------------------
__global__ __launch_bounds__(256) void att_kernel(const float* __restrict__ CB,
                                                  const float* __restrict__ AW,
                                                  const int* __restrict__ idx,
                                                  float* __restrict__ att) {
  const int tid = threadIdx.x;
  const int lane = tid & 31;
  const int wv = tid >> 5;
  const int token = blockIdx.x * 8 + wv;        // BL = 6272 * 8 exactly
  const int e = idx[token];
  const float* cb = CB + (size_t)e * CC;
  float s = 0.f;
  for (int c = lane; c < CC; c += 32) s += cb[c] * AW[c * 3 + 1];  // center tap
#pragma unroll
  for (int off = 16; off >= 1; off >>= 1) s += __shfl_xor(s, off, 32);
  if (lane == 0) att[token] = fmaxf(s, 0.f);
}

// ---------------------------------------------------------------------------
// Kernel 5: per-batch softmax over L, then out = q + x*mask; idx -> float tail
// ---------------------------------------------------------------------------
__global__ __launch_bounds__(256) void softmax_out_kernel(const float* __restrict__ X,
                                                          const float* __restrict__ CB,
                                                          const int* __restrict__ idx,
                                                          const float* __restrict__ att,
                                                          float* __restrict__ out,
                                                          float* __restrict__ idx_out_f) {
  __shared__ float sm[LL];
  __shared__ int   si[LL];
  __shared__ float red[256];
  const int b = blockIdx.x;
  const int tid = threadIdx.x;

  for (int l = tid; l < LL; l += 256) {
    sm[l] = att[b * LL + l];
    si[l] = idx[b * LL + l];
  }
  __syncthreads();

  // max over L
  float m = -3.0e38f;
  for (int l = tid; l < LL; l += 256) m = fmaxf(m, sm[l]);
  red[tid] = m; __syncthreads();
  for (int st = 128; st > 0; st >>= 1) {
    if (tid < st) red[tid] = fmaxf(red[tid], red[tid + st]);
    __syncthreads();
  }
  m = red[0];
  __syncthreads();

  // sum of exp
  float s = 0.f;
  for (int l = tid; l < LL; l += 256) s += __expf(sm[l] - m);
  red[tid] = s; __syncthreads();
  for (int st = 128; st > 0; st >>= 1) {
    if (tid < st) red[tid] += red[tid + st];
    __syncthreads();
  }
  const float inv = 1.f / red[0];
  __syncthreads();

  for (int l = tid; l < LL; l += 256) sm[l] = __expf(sm[l] - m) * inv;
  __syncthreads();

  // out[b,l,:] = codebook[idx] + x * mask  (coalesced along C)
  const size_t base = (size_t)b * LL * CC;
  for (int l = 0; l < LL; ++l) {
    const float msk = sm[l];
    const float* cb = CB + (size_t)si[l] * CC;
    const size_t rb = base + (size_t)l * CC;
    for (int c = tid; c < CC; c += 256)
      out[rb + c] = cb[c] + X[rb + c] * msk;
  }
  for (int l = tid; l < LL; l += 256)
    idx_out_f[b * LL + l] = (float)si[l];
}

// ---------------------------------------------------------------------------
// Host launcher
// ---------------------------------------------------------------------------
extern "C" void kernel_launch(void* const* d_in, const int* in_sizes, int n_in,
                              void* d_out, int out_size, void* d_ws, size_t ws_size,
                              hipStream_t stream) {
  (void)in_sizes; (void)n_in; (void)out_size; (void)ws_size;
  const float* X  = (const float*)d_in[0];   // [B,L,C]
  const float* CB = (const float*)d_in[1];   // [K,C]
  const float* AW = (const float*)d_in[2];   // [1,C,3]
  float* out = (float*)d_out;
  float* idx_out_f = out + (size_t)BL * CC;

  uint8_t* ws = (uint8_t*)d_ws;
  uint16_t* P   = (uint16_t*)ws;                              // 3,145,728 B packed bf16 codebook
  float*    hn  = (float*)(ws + 3145728);                     // 8 KB half-norms
  int*      idb = (int*)(ws + 3145728 + 8192);                // 50176 * 4 B indices
  float*    atb = (float*)(ws + 3145728 + 8192 + 200704);     // 50176 * 4 B attention logits

  hipLaunchKernelGGL(halfnorm_kernel, dim3(KK), dim3(256), 0, stream, CB, hn);
  hipLaunchKernelGGL(pack_kernel, dim3(NTILE * NCHUNK), dim3(32), 0, stream, CB, P);
  hipLaunchKernelGGL(vq_argmax_kernel, dim3(BL / ROWS_PER_BLOCK), dim3(256), 0, stream,
                     X, P, hn, idb);
  hipLaunchKernelGGL(att_kernel, dim3(BL / 8), dim3(256), 0, stream, CB, AW, idb, atb);
  hipLaunchKernelGGL(softmax_out_kernel, dim3(BB), dim3(256), 0, stream,
                     X, CB, idb, atb, out, idx_out_f);
}
